// PytorchCLER_28887950033476
// MI455X (gfx1250) — matrix operand order repaired and checked by reading, not compiled, not run on hardware
//
#include <hip/hip_runtime.h>
#include <hip/hip_bf16.h>

typedef __attribute__((ext_vector_type(16))) _Float16     v16h;
typedef __attribute__((ext_vector_type(4)))  _Float16     v4h;
typedef __attribute__((ext_vector_type(8)))  float        v8f;
typedef __attribute__((ext_vector_type(4)))  unsigned int v4u;

#define B_N      8192
#define DIM_D    128
#define P_N      16384
#define TEMP_INV 10.0f
#define ROWBLOCKS 64   // 8192 / 128
#define COLSPLITS 16   // each block handles 4 col tiles of 128

union ABFrag { v4u q[2]; v16h h; };

// ---------------- workspace layout (bytes) ----------------
static constexpr size_t OFF_UI   = 0;                                    // [B,128] f32
static constexpr size_t OFF_E    = OFF_UI   + (size_t)B_N * DIM_D * 4;   // [B,128] f32
static constexpr size_t OFF_UH   = OFF_E    + (size_t)B_N * DIM_D * 4;   // [B,128] f16
static constexpr size_t OFF_EH   = OFF_UH   + (size_t)B_N * DIM_D * 2;   // [B,128] f16
static constexpr size_t OFF_COLP = OFF_EH   + (size_t)B_N * DIM_D * 2;   // [64,B] f32
static constexpr size_t OFF_ROWP = OFF_COLP + (size_t)ROWBLOCKS * B_N*4; // [16,B] f32
static constexpr size_t OFF_LSEC = OFF_ROWP + (size_t)COLSPLITS * B_N*4; // [B] f32
static constexpr size_t OFF_LSER = OFF_LSEC + (size_t)B_N * 4;           // [B] f32
static constexpr size_t OFF_ESUM = OFF_LSER + (size_t)B_N * 4;           // [128] f32
static constexpr size_t OFF_S    = OFF_ESUM + 512;                       // scalar
static constexpr size_t OFF_MASK = OFF_S    + 256;                       // [B,256] u32
static constexpr size_t OFF_TERM = OFF_MASK + (size_t)B_N * (B_N/32)*4;  // [B] f32

// ---- LDS layout inside dynamic smem (byte offsets from dynamic base) ----
#define LDS_A    0u          // 128 x 64 dwords (f16 pairs) = 32KB
#define LDS_E0   32768u      // 32KB, buffer 0
#define LDS_E1   65536u      // 32KB, buffer 1
#define LDS_SUMS 98304u      // rowsum[128] + colsum[128]
#define LDS_TOTAL (98304u + 1024u)

// Issue one 32KB tile copy (global -> LDS) as 8 async b128 transfers per thread.
// GVS addressing: 64-bit SGPR base + 32-bit VGPR byte offset. Tracked by ASYNCcnt.
__device__ __forceinline__ void async_tile_issue(unsigned lds_base, const _Float16* gbase, int tid) {
  unsigned long long g = (unsigned long long)gbase;
#pragma unroll
  for (int i = 0; i < 8; ++i) {
    unsigned boff = (unsigned)(i * 256 + tid) * 16u;
    asm volatile("global_load_async_to_lds_b128 %0, %1, %2"
                 :: "v"(lds_base + boff), "v"(boff), "s"(g) : "memory");
  }
}

// ---------------- 1) gather + mix + normalize + f16 copies ----------------
__global__ void prep_kernel(const float* __restrict__ user_t, const float* __restrict__ item_t,
                            const float* __restrict__ exp_t,
                            const int* __restrict__ uid, const int* __restrict__ iid,
                            const int* __restrict__ eid,
                            float* __restrict__ UI, float* __restrict__ E,
                            _Float16* __restrict__ UH, _Float16* __restrict__ EH) {
  int lane = threadIdx.x & 31;
  int row  = blockIdx.x * (blockDim.x >> 5) + (threadIdx.x >> 5);
  if (row >= B_N) return;
  int u = uid[row], it = iid[row], ex = eid[row];
  float4 uv = *(const float4*)(user_t + (size_t)u  * DIM_D + lane * 4);
  float4 iv = *(const float4*)(item_t + (size_t)it * DIM_D + lane * 4);
  float4 ev = *(const float4*)(exp_t  + (size_t)ex * DIM_D + lane * 4);
  float4 m;
  m.x = 0.5f*uv.x + 0.5f*iv.x; m.y = 0.5f*uv.y + 0.5f*iv.y;
  m.z = 0.5f*uv.z + 0.5f*iv.z; m.w = 0.5f*uv.w + 0.5f*iv.w;
  float ssu = m.x*m.x + m.y*m.y + m.z*m.z + m.w*m.w;
  float sse = ev.x*ev.x + ev.y*ev.y + ev.z*ev.z + ev.w*ev.w;
  for (int o = 1; o < 32; o <<= 1) { ssu += __shfl_xor(ssu, o, 32); sse += __shfl_xor(sse, o, 32); }
  float su = 1.0f / fmaxf(sqrtf(ssu), 1e-12f);
  float se = 1.0f / fmaxf(sqrtf(sse), 1e-12f);
  m.x *= su; m.y *= su; m.z *= su; m.w *= su;
  ev.x *= se; ev.y *= se; ev.z *= se; ev.w *= se;
  *(float4*)(UI + (size_t)row * DIM_D + lane * 4) = m;
  *(float4*)(E  + (size_t)row * DIM_D + lane * 4) = ev;
  v4h mh; mh.x = (_Float16)m.x;  mh.y = (_Float16)m.y;  mh.z = (_Float16)m.z;  mh.w = (_Float16)m.w;
  v4h eh; eh.x = (_Float16)ev.x; eh.y = (_Float16)ev.y; eh.z = (_Float16)ev.z; eh.w = (_Float16)ev.w;
  *(v4h*)(UH + (size_t)row * DIM_D + lane * 4) = mh;
  *(v4h*)(EH + (size_t)row * DIM_D + lane * 4) = eh;
}

// ---------------- 2) fused f16-WMMA GEMM + exp + row/col partial sums ----------------
// grid = 1024 blocks (64 rowblocks x 16 colsplits), 256 threads (8 waves as 4x2).
// A tile + double-buffered E tiles streamed via GLOBAL_LOAD_ASYNC_TO_LDS_B128.
__global__ __launch_bounds__(256) void gemm_stats_kernel(
    const _Float16* __restrict__ UH, const _Float16* __restrict__ EH,
    float* __restrict__ ColPart, float* __restrict__ RowPart) {
  extern __shared__ char smem[];
  unsigned int* At = (unsigned int*)(smem + LDS_A);
  float* rowsum = (float*)(smem + LDS_SUMS);     // 128
  float* colsum = rowsum + 128;                  // 128

  const unsigned ldsbase = __builtin_amdgcn_groupstaticsize();  // dynamic LDS starts here
  const int bid = blockIdx.x;
  const int rb  = bid >> 4;     // 0..63
  const int cs  = bid & 15;     // 0..15
  const int tid = threadIdx.x;
  const int lane = tid & 31;
  const int wv  = tid >> 5;     // 0..7
  const int wm  = (wv >> 1) * 32;   // wave row base in tile
  const int wn  = (wv & 1) * 64;    // wave col base in tile
  const int half = lane >> 4;       // 0 or 1

  // kick off async staging: A tile, then E tile 0 (ASYNCcnt completes in order)
  async_tile_issue(ldsbase + LDS_A,  UH + (size_t)rb * 128 * DIM_D, tid);
  async_tile_issue(ldsbase + LDS_E0, EH + (size_t)(cs * 4) * 128 * DIM_D, tid);
  if (tid < 128) rowsum[tid] = 0.0f;

  for (int jt = 0; jt < 4; ++jt) {
    const int cur = jt & 1;
    const int j0  = (cs * 4 + jt) * 128;
    if (jt < 3)   // stream next E tile into the other buffer while we compute
      async_tile_issue(ldsbase + LDS_E0 + (unsigned)(1 - cur) * 32768u,
                       EH + (size_t)(j0 + 128) * DIM_D, tid);
    if (tid < 128) colsum[tid] = 0.0f;
    // drain everything except the 8 just-issued next-tile transfers
    if (jt < 3) asm volatile("s_wait_asynccnt 0x8" ::: "memory");
    else        asm volatile("s_wait_asynccnt 0x0" ::: "memory");
    __syncthreads();

    const unsigned int* Et = (const unsigned int*)(smem + LDS_E0 + (unsigned)cur * 32768u);
    v8f acc[2][4];
    {
      v8f z = {0.f,0.f,0.f,0.f,0.f,0.f,0.f,0.f};
      for (int mt = 0; mt < 2; ++mt)
        for (int nt = 0; nt < 4; ++nt) acc[mt][nt] = z;
    }

    // K loop: 128 = 4 chunks of 32
    for (int kc = 0; kc < 4; ++kc) {
      ABFrag fa[2];
      // A fragment, 16-bit 16x32 layout (ISA 7.12.2): lanes<16 -> K{0..7,16..23}, lanes>=16 -> K{8..15,24..31}
      for (int mt = 0; mt < 2; ++mt) {
        int m = wm + mt * 16 + (lane & 15);
        const unsigned int* p = At + m * 64 + kc * 16 + (half << 2);
        fa[mt].q[0] = *(const v4u*)p;
        fa[mt].q[1] = *(const v4u*)(p + 8);
      }
      for (int nt = 0; nt < 4; ++nt) {
        // B fragment 32x16: N = lane%16, lanes<16 -> K{0..7,16..23}, lanes>=16 -> K{8..15,24..31}
        ABFrag fb;
        int n = wn + nt * 16 + (lane & 15);
        const unsigned int* p = Et + n * 64 + kc * 16 + (half << 2);
        fb.q[0] = *(const v4u*)p;
        fb.q[1] = *(const v4u*)(p + 8);
        acc[0][nt] = __builtin_amdgcn_wmma_f32_16x16x32_f16(false, fa[0].h, false, fb.h,
                                                            (short)0, acc[0][nt], false, false);
        acc[1][nt] = __builtin_amdgcn_wmma_f32_16x16x32_f16(false, fa[1].h, false, fb.h,
                                                            (short)0, acc[1][nt], false, false);
      }
    }

    // exp + reductions. C layout: VGPR r, lanes<16 -> (M=r,     N=lane),
    //                                     lanes>=16 -> (M=r+8, N=lane-16)
    for (int mt = 0; mt < 2; ++mt) {
      float ex[4][8];
      float colp[4];
      for (int nt = 0; nt < 4; ++nt) {
        float cp = 0.0f;
        for (int r = 0; r < 8; ++r) {
          float e = __expf(acc[mt][nt][r] * TEMP_INV);
          ex[nt][r] = e; cp += e;
        }
        colp[nt] = cp;
      }
      for (int nt = 0; nt < 4; ++nt) {                 // column sums over this wave's 32 rows
        float c = colp[nt] + __shfl_xor(colp[nt], 16, 32);
        if (lane < 16) atomicAdd(&colsum[wn + nt * 16 + lane], c);
      }
      for (int r = 0; r < 8; ++r) {                    // row sums over this wave's 64 cols
        float rp = ex[0][r] + ex[1][r] + ex[2][r] + ex[3][r];
        rp += __shfl_xor(rp, 1, 32);
        rp += __shfl_xor(rp, 2, 32);
        rp += __shfl_xor(rp, 4, 32);
        rp += __shfl_xor(rp, 8, 32);
        if ((lane & 15) == 0)
          atomicAdd(&rowsum[wm + mt * 16 + r + (half << 3)], rp);
      }
    }
    __syncthreads();
    if (tid < 128) ColPart[(size_t)rb * B_N + j0 + tid] = colsum[tid];
  }
  if (tid < 128) RowPart[(size_t)cs * B_N + rb * 128 + tid] = rowsum[tid];
}

// ---------------- 3) deterministic reductions ----------------
__global__ void col_reduce_kernel(const float* __restrict__ ColPart, float* __restrict__ LSEc) {
  int j = blockIdx.x * blockDim.x + threadIdx.x;
  if (j >= B_N) return;
  float s = 0.0f;
  for (int rb = 0; rb < ROWBLOCKS; ++rb) s += ColPart[(size_t)rb * B_N + j];
  LSEc[j] = __logf(s);
}
__global__ void row_reduce_kernel(const float* __restrict__ RowPart, float* __restrict__ LSEr) {
  int i = blockIdx.x * blockDim.x + threadIdx.x;
  if (i >= B_N) return;
  float s = 0.0f;
  for (int c = 0; c < COLSPLITS; ++c) s += RowPart[(size_t)c * B_N + i];
  LSEr[i] = __logf(s);
}
__global__ void sum_lsec_kernel(const float* __restrict__ LSEc, float* __restrict__ S) {
  __shared__ float sh[256];
  float s = 0.0f;
  for (int j = threadIdx.x; j < B_N; j += 256) s += LSEc[j];
  sh[threadIdx.x] = s; __syncthreads();
  for (int o = 128; o > 0; o >>= 1) { if (threadIdx.x < o) sh[threadIdx.x] += sh[threadIdx.x + o]; __syncthreads(); }
  if (threadIdx.x == 0) S[0] = sh[0];
}
__global__ void esum_kernel(const float* __restrict__ E, float* __restrict__ Esum) {
  int d = threadIdx.x;  // 128 threads
  float s = 0.0f;
  for (int i = 0; i < B_N; ++i) s += E[(size_t)i * DIM_D + d];
  Esum[d] = s;
}

// ---------------- 4) positive-cell mask ----------------
__global__ void zero_mask_kernel(unsigned int* __restrict__ m, unsigned int n) {
  unsigned int i = blockIdx.x * blockDim.x + threadIdx.x;
  for (; i < n; i += gridDim.x * blockDim.x) m[i] = 0u;
}
__global__ void mask_set_kernel(const int* __restrict__ pos, unsigned int* __restrict__ mask) {
  int t = blockIdx.x * blockDim.x + threadIdx.x;
  if (t >= P_N + B_N) return;
  int r, c;
  if (t < P_N) { r = pos[2 * t]; c = pos[2 * t + 1]; } else { r = t - P_N; c = r; }
  atomicOr(&mask[(size_t)r * (B_N / 32) + (c >> 5)], 1u << (c & 31));
}

// ---------------- 5) per-row loss terms ----------------
__global__ void row_loss_kernel(const unsigned int* __restrict__ mask,
                                const float* __restrict__ UI, const float* __restrict__ E,
                                const float* __restrict__ Esum, const float* __restrict__ LSEc,
                                const float* __restrict__ LSEr, const float* __restrict__ Sp,
                                float* __restrict__ term) {
  int i = blockIdx.x * blockDim.x + threadIdx.x;
  if (i >= B_N) return;
  const float* ui = UI + (size_t)i * DIM_D;
  float S = Sp[0], lser = LSEr[i];
  int npos = 0; float spdot = 0.0f, splsec = 0.0f;
  for (int w = 0; w < B_N / 32; ++w) {
    unsigned int m = mask[(size_t)i * (B_N / 32) + w];
    while (m) {
      int b = __ffs(m) - 1; m &= m - 1;
      int j = w * 32 + b;
      ++npos; splsec += LSEc[j];
      const float* ej = E + (size_t)j * DIM_D;
      float d = 0.0f;
      for (int k = 0; k < DIM_D; ++k) d += ui[k] * ej[k];
      spdot += d;
    }
  }
  float rss = 0.0f;
  for (int k = 0; k < DIM_D; ++k) rss += ui[k] * Esum[k];
  rss *= TEMP_INV;
  float loss_pos = 2.0f * TEMP_INV * spdot - (float)npos * lser - splsec;
  float total    = 2.0f * rss - (float)B_N * lser - S;
  float loss_neg = total - loss_pos;
  term[i] = loss_pos / (float)npos + loss_neg / (float)(B_N - npos);
}

__global__ void final_reduce_kernel(const float* __restrict__ term, float* __restrict__ out) {
  __shared__ float sh[256];
  float s = 0.0f;
  for (int i = threadIdx.x; i < B_N; i += 256) s += term[i];
  sh[threadIdx.x] = s; __syncthreads();
  for (int o = 128; o > 0; o >>= 1) { if (threadIdx.x < o) sh[threadIdx.x] += sh[threadIdx.x + o]; __syncthreads(); }
  if (threadIdx.x == 0) out[0] = -(sh[0] / (float)B_N);
}

// ---------------- launch ----------------
extern "C" void kernel_launch(void* const* d_in, const int* in_sizes, int n_in,
                              void* d_out, int out_size, void* d_ws, size_t ws_size,
                              hipStream_t stream) {
  const float* user_t = (const float*)d_in[0];
  const float* item_t = (const float*)d_in[1];
  const float* exp_t  = (const float*)d_in[2];
  const int*   uid    = (const int*)d_in[3];
  const int*   iid    = (const int*)d_in[4];
  const int*   eid    = (const int*)d_in[5];
  const int*   pos    = (const int*)d_in[6];

  char* ws = (char*)d_ws;
  float*        UI   = (float*)(ws + OFF_UI);
  float*        E    = (float*)(ws + OFF_E);
  _Float16*     UH   = (_Float16*)(ws + OFF_UH);
  _Float16*     EH   = (_Float16*)(ws + OFF_EH);
  float*        ColP = (float*)(ws + OFF_COLP);
  float*        RowP = (float*)(ws + OFF_ROWP);
  float*        LSEc = (float*)(ws + OFF_LSEC);
  float*        LSEr = (float*)(ws + OFF_LSER);
  float*        Esum = (float*)(ws + OFF_ESUM);
  float*        S    = (float*)(ws + OFF_S);
  unsigned int* mask = (unsigned int*)(ws + OFF_MASK);
  float*        term = (float*)(ws + OFF_TERM);

  prep_kernel<<<B_N / 8, 256, 0, stream>>>(user_t, item_t, exp_t, uid, iid, eid, UI, E, UH, EH);

  gemm_stats_kernel<<<ROWBLOCKS * COLSPLITS, 256, LDS_TOTAL, stream>>>(UH, EH, ColP, RowP);

  col_reduce_kernel<<<B_N / 256, 256, 0, stream>>>(ColP, LSEc);
  row_reduce_kernel<<<B_N / 256, 256, 0, stream>>>(RowP, LSEr);
  sum_lsec_kernel<<<1, 256, 0, stream>>>(LSEc, S);
  esum_kernel<<<1, 128, 0, stream>>>(E, Esum);

  zero_mask_kernel<<<2048, 256, 0, stream>>>(mask, (unsigned int)(B_N * (B_N / 32)));
  mask_set_kernel<<<(P_N + B_N + 255) / 256, 256, 0, stream>>>(pos, mask);

  row_loss_kernel<<<B_N / 256, 256, 0, stream>>>(mask, UI, E, Esum, LSEc, LSEr, S, term);
  final_reduce_kernel<<<1, 256, 0, stream>>>(term, (float*)d_out);
}
